// Decoder_40690520162537
// MI455X (gfx1250) — matrix operand order repaired and checked by reading, not compiled, not run on hardware
//
#include <hip/hip_runtime.h>
#include <hip/hip_bf16.h>

typedef __attribute__((ext_vector_type(16))) _Float16 v16h;
typedef __attribute__((ext_vector_type(8)))  float    v8f;
typedef __attribute__((ext_vector_type(4)))  unsigned int v4u;
typedef __attribute__((ext_vector_type(8)))  unsigned int v8u;

#define BB 8
#define CC 128
#define HH 56
#define WW 56
#define HWP (HH * WW)
#define NPIX (BB * HWP)

// ---------------------------------------------------------------------------
// Tap-major weight pack: out[o*Kp + r*Cin + ci] = w[(o*Cin + ci)*9 + r],
// zero-padding rows o >= Mvalid (tile alignment). Kp = Cin*9.
__global__ void pack_tapmajor_kernel(const float* __restrict__ w, _Float16* __restrict__ outp,
                                     int Mvalid, int Mp, int Cin) {
    const int Kp = Cin * 9;
    long i = (long)blockIdx.x * blockDim.x + threadIdx.x;
    if (i >= (long)Mp * Kp) return;
    int o = (int)(i / Kp);
    int k = (int)(i - (long)o * Kp);
    int r = k / Cin;
    int c = k - r * Cin;
    outp[i] = (o < Mvalid) ? (_Float16)w[((long)o * Cin + c) * 9 + r] : (_Float16)0.0f;
}

// ---------------------------------------------------------------------------
// Implicit-GEMM 3x3 SAME conv via WMMA f16->f32 (tap-major K order).
//   D[M=Co, N=B*H*W] = Wpack[M, Kp] * Bpanel[Kp, N] + bias,  Kp = Cin*9
// Block tile: 128 M x 64 N, K stepped by 64 per barrier pair.
// 256 threads = 8 waves: wave w -> mSub = w>>1 (32 rows), nSub = w&1 (32 px).
// Each wave: 2x2 register blocking of 16x16 WMMA tiles -> 8 WMMAs per
// sync iteration (2 K-steps x 2 M-frags x 2 N-frags).
// Tap-major => each 64-K block lies in a single tap r: (ky,kx) are uniform,
// the boundary check is per-block, and the 16 gathered channels per thread
// are a constant-stride (HW) clause of loads.
// MODE 0: im2col on the fly from fp32 NCHW (channel split x0/x1 for concat).
// MODE 1: pre-materialized f16 [N, Kp] panel (deform patches).
template <int MODE>
__global__ __launch_bounds__(256)
void conv_gemm_kernel(const float* __restrict__ x0, const float* __restrict__ x1, int Csplit,
                      const _Float16* __restrict__ bmat,
                      const _Float16* __restrict__ wpack, const float* __restrict__ bias,
                      float* __restrict__ outp, int H, int W, int Co, int Kp) {
    __shared__ __align__(16) _Float16 As[128 * 64];  // 128 M rows x 64 K  (16 KB)
    __shared__ __align__(16) _Float16 Bs[64 * 64];   //  64 N cols x 64 K  ( 8 KB)

    const int tid  = threadIdx.x;
    const int wave = tid >> 5;
    const int lane = tid & 31;
    const int l15  = lane & 15;
    const int mSub = wave >> 1;      // 0..3 -> 32 rows each
    const int nSub = wave & 1;       // 0..1 -> 32 px each
    const int mBase = blockIdx.x * 128;
    const int nBase = blockIdx.y * 64;
    const int HW = H * W;
    const int kp4 = Kp >> 3;         // Kp in uint4 units
    const int Cin = Kp / 9;

    // im2col mapping for MODE 0: thread owns pixel nPix, channel segment kseg..+15
    const int nPix = tid & 63;              // 0..63 (coalesced across a wave)
    const int kseg = (tid >> 6) * 16;       // 0,16,32,48
    int b_g = 0, oy_g = 0, ox_g = 0;
    if (MODE == 0) {
        int pn = nBase + nPix;
        b_g = pn / HW;
        int p = pn - b_g * HW;
        oy_g = p / W;
        ox_g = p - oy_g * W;
    }

    const v4u* wq4 = (const v4u*)wpack;

    v8f acc00 = {}, acc01 = {}, acc10 = {}, acc11 = {};

    const int kOffA = (lane < 16) ? 0 : 8;    // A 16x32 layout
    const int kOffB = (lane < 16) ? 0 : 8;    // B 32x16 layout
    const int arow0 = mSub * 32 + l15;
    const int arow1 = arow0 + 16;
    const int bcol0 = nSub * 32 + l15;
    const int bcol1 = bcol0 + 16;

    auto fragA = [&](int row, int ks) -> v16h {
        v4u a0 = *(const v4u*)(As + row * 64 + ks + kOffA);
        v4u a1 = *(const v4u*)(As + row * 64 + ks + 16 + kOffA);
        v8u au = __builtin_shufflevector(a0, a1, 0, 1, 2, 3, 4, 5, 6, 7);
        return __builtin_bit_cast(v16h, au);
    };
    auto fragB = [&](int col, int ks) -> v16h {
        v4u b0 = *(const v4u*)(Bs + col * 64 + ks + kOffB);
        v4u b1 = *(const v4u*)(Bs + col * 64 + ks + kOffB + 16);
        v8u bu = __builtin_shufflevector(b0, b1, 0, 1, 2, 3, 4, 5, 6, 7);
        return __builtin_bit_cast(v16h, bu);
    };

    for (int kb = 0; kb < Kp; kb += 64) {
        // ---- stage A (weights): 128x64 halves = 1024 uint4; 4 per thread
        v4u wv[4];
        #pragma unroll
        for (int i = 0; i < 4; ++i) {
            int t = tid + i * 256;
            wv[i] = wq4[(long)(mBase + (t >> 3)) * kp4 + (kb >> 3) + (t & 7)];
        }
        if (kb + 64 < Kp) {
            __builtin_prefetch((const void*)(wq4 + (long)(mBase + (tid >> 3)) * kp4 + ((kb + 64) >> 3) + (tid & 7)), 0, 1);
            __builtin_prefetch((const void*)(wq4 + (long)(mBase + ((tid + 512) >> 3)) * kp4 + ((kb + 64) >> 3) + (tid & 7)), 0, 1);
        }

        // ---- stage B: gather/copy into registers first, then LDS
        if (MODE == 0) {
            // Whole 64-K block lies in one tap: uniform (ky,kx), per-block validity.
            const int rTap = kb / Cin;                 // uniform
            const int cBlk = kb - rTap * Cin;          // uniform, multiple of 64
            const int ky = rTap / 3 - 1;
            const int kx = rTap - (rTap / 3) * 3 - 1;
            const int iy = oy_g + ky, ix = ox_g + kx;  // per-thread, per-block
            const bool valid = (iy >= 0) & (iy < H) & (ix >= 0) & (ix < W);
            const int c0 = cBlk + kseg;                // first channel of this segment
            const float* src = (c0 < Csplit) ? x0 : x1;
            const int cl0 = (c0 < Csplit) ? c0 : (c0 - Csplit);
            const float* base = src + ((long)(b_g * Csplit + cl0)) * HW + iy * W + ix;

            float vals[16];
            if (valid) {
                #pragma unroll
                for (int j = 0; j < 16; ++j) vals[j] = base[j * HW];
            } else {
                #pragma unroll
                for (int j = 0; j < 16; ++j) vals[j] = 0.0f;
            }
            _Float16 hv[16] __attribute__((aligned(16)));
            #pragma unroll
            for (int j = 0; j < 16; ++j) hv[j] = (_Float16)vals[j];
            v4u q0, q1;
            __builtin_memcpy(&q0, hv, 16);
            __builtin_memcpy(&q1, hv + 8, 16);
            __syncthreads();  // previous iteration's compute done
            #pragma unroll
            for (int i = 0; i < 4; ++i) *(v4u*)(As + (tid + i * 256) * 8) = wv[i];
            *(v4u*)(Bs + nPix * 64 + kseg)     = q0;
            *(v4u*)(Bs + nPix * 64 + kseg + 8) = q1;
        } else {
            const v4u* bq4 = (const v4u*)bmat;
            v4u ba = bq4[(long)(nBase + (tid >> 3)) * kp4 + (kb >> 3) + (tid & 7)];
            v4u bb = bq4[(long)(nBase + ((tid + 256) >> 3)) * kp4 + (kb >> 3) + (tid & 7)];
            __syncthreads();  // previous iteration's compute done
            #pragma unroll
            for (int i = 0; i < 4; ++i) *(v4u*)(As + (tid + i * 256) * 8) = wv[i];
            *(v4u*)(Bs + tid * 8)         = ba;
            *(v4u*)(Bs + (tid + 256) * 8) = bb;
        }
        __syncthreads();

        // ---- compute: 2 K-steps of 32, 2x2 register blocking (8 WMMAs)
        #pragma unroll
        for (int ks = 0; ks < 64; ks += 32) {
            v16h ah0 = fragA(arow0, ks);
            v16h ah1 = fragA(arow1, ks);
            v16h bh0 = fragB(bcol0, ks);
            v16h bh1 = fragB(bcol1, ks);
            acc00 = __builtin_amdgcn_wmma_f32_16x16x32_f16(false, ah0, false, bh0, (short)0, acc00, false, false);
            acc01 = __builtin_amdgcn_wmma_f32_16x16x32_f16(false, ah0, false, bh1, (short)0, acc01, false, false);
            acc10 = __builtin_amdgcn_wmma_f32_16x16x32_f16(false, ah1, false, bh0, (short)0, acc10, false, false);
            acc11 = __builtin_amdgcn_wmma_f32_16x16x32_f16(false, ah1, false, bh1, (short)0, acc11, false, false);
        }
    }

    // ---- epilogue: C/D layout — lanes 0-15: N=lane, M=g; lanes 16-31: N=lane-16, M=8+g
    const int mHalf = (lane >> 4) * 8;
    #pragma unroll
    for (int mm = 0; mm < 2; ++mm) {
        #pragma unroll
        for (int nn = 0; nn < 2; ++nn) {
            const int pn = nBase + nSub * 32 + nn * 16 + l15;
            const int b  = pn / HW;
            const int p  = pn - b * HW;
            const v8f acc = mm ? (nn ? acc11 : acc10) : (nn ? acc01 : acc00);
            #pragma unroll
            for (int g = 0; g < 8; ++g) {
                int co = mBase + mSub * 32 + mm * 16 + mHalf + g;
                if (co < Co) {
                    outp[((long)(b * Co + co)) * HW + p] = acc[g] + bias[co];
                }
            }
        }
    }
}

// ---------------------------------------------------------------------------
// Deformable bilinear sampling: produces f16 patch panel samp[pixel][r*C + c].
__global__ __launch_bounds__(128)
void deform_sample_kernel(const float* __restrict__ h1, const float* __restrict__ off,
                          _Float16* __restrict__ samp, int C, int H, int W) {
    const int pn = blockIdx.x;       // pixel
    const int r  = blockIdx.y;       // tap 0..8
    const int c  = threadIdx.x;      // channel
    const int HW = H * W;
    const int b  = pn / HW;
    const int p  = pn - b * HW;
    const int oy = p / W;
    const int ox = p - oy * W;

    float offy = off[((long)(b * 18 + 2 * r)) * HW + p];
    float offx = off[((long)(b * 18 + 2 * r + 1)) * HW + p];
    int ky = r / 3 - 1;
    int kx = r - (r / 3) * 3 - 1;
    float py = (float)oy + (float)ky + offy;
    float px = (float)ox + (float)kx + offx;
    float y0 = floorf(py), x0f = floorf(px);
    float wy1 = py - y0, wy0 = 1.0f - wy1;
    float wx1 = px - x0f, wx0 = 1.0f - wx1;

    const float* base = h1 + ((long)(b * C + c)) * HW;
    float acc = 0.0f;
    #pragma unroll
    for (int dy = 0; dy < 2; ++dy) {
        #pragma unroll
        for (int dx = 0; dx < 2; ++dx) {
            float yc = y0 + (float)dy;
            float xc = x0f + (float)dx;
            bool valid = (yc >= 0.0f) && (yc <= (float)(H - 1)) &&
                         (xc >= 0.0f) && (xc <= (float)(W - 1));
            int iy = (int)fminf(fmaxf(yc, 0.0f), (float)(H - 1));
            int ix = (int)fminf(fmaxf(xc, 0.0f), (float)(W - 1));
            float wt = (dy ? wy1 : wy0) * (dx ? wx1 : wx0);
            float v = base[iy * W + ix];
            acc += valid ? v * wt : 0.0f;
        }
    }
    samp[(long)pn * (9 * C) + r * C + c] = (_Float16)acc;
}

// ---------------------------------------------------------------------------
// BatchNorm statistics: one block per channel, reduce over B*H*W.
__global__ __launch_bounds__(256)
void bn_stats_kernel(const float* __restrict__ h, float* __restrict__ stats,
                     int B, int C, int HW) {
    const int c = blockIdx.x;
    const int tid = threadIdx.x;
    const int total = B * HW;
    float s = 0.0f, s2 = 0.0f;
    for (int i = tid; i < total; i += blockDim.x) {
        int b = i / HW;
        int p = i - b * HW;
        float v = h[((long)(b * C + c)) * HW + p];
        s += v;
        s2 += v * v;
    }
    __shared__ float r1[256];
    __shared__ float r2[256];
    r1[tid] = s;
    r2[tid] = s2;
    __syncthreads();
    for (int k = 128; k > 0; k >>= 1) {
        if (tid < k) { r1[tid] += r1[tid + k]; r2[tid] += r2[tid + k]; }
        __syncthreads();
    }
    if (tid == 0) {
        float inv = 1.0f / (float)total;
        float mean = r1[0] * inv;
        float var = r2[0] * inv - mean * mean;
        stats[c] = mean;
        stats[C + c] = rsqrtf(var + 1e-5f);
    }
}

// Fused BN apply + exact GELU -> d_out (h output).
__global__ __launch_bounds__(256)
void bn_gelu_kernel(const float* __restrict__ h, const float* __restrict__ stats,
                    const float* __restrict__ gamma, const float* __restrict__ beta,
                    float* __restrict__ outp, int C, int HW, long total) {
    long i = (long)blockIdx.x * blockDim.x + threadIdx.x;
    if (i >= total) return;
    int c = (int)((i / HW) % C);
    float v = h[i];
    v = gamma[c] * (v - stats[c]) * stats[C + c] + beta[c];
    float g = 0.5f * v * (1.0f + erff(v * 0.70710678118654752f));
    outp[i] = g;
}

// ---------------------------------------------------------------------------
extern "C" void kernel_launch(void* const* d_in, const int* in_sizes, int n_in,
                              void* d_out, int out_size, void* d_ws, size_t ws_size,
                              hipStream_t stream) {
    (void)in_sizes; (void)n_in; (void)out_size; (void)ws_size;

    const float* x       = (const float*)d_in[0];
    const float* y       = (const float*)d_in[1];
    const float* conv_w  = (const float*)d_in[2];
    const float* conv_b  = (const float*)d_in[3];
    const float* off_w   = (const float*)d_in[4];
    const float* off_b   = (const float*)d_in[5];
    const float* dcn_w   = (const float*)d_in[6];
    const float* dcn_b   = (const float*)d_in[7];
    const float* conv2_w = (const float*)d_in[8];
    const float* conv2_b = (const float*)d_in[9];
    const float* bn_g    = (const float*)d_in[10];
    const float* bn_b    = (const float*)d_in[11];
    const float* mask_w  = (const float*)d_in[12];
    const float* mask_b  = (const float*)d_in[13];

    float* out_h    = (float*)d_out;
    float* out_mask = out_h + (long)BB * CC * HWP;

    // ---- workspace carve-out (256B aligned)
    char* wsp = (char*)d_ws;
    size_t off_bytes = 0;
    auto take = [&](size_t bytes) -> void* {
        void* p = wsp + off_bytes;
        off_bytes = (off_bytes + bytes + 255) & ~(size_t)255;
        return p;
    };
    const int K1 = 2304;  // 9 * 2C  (tap-major: k = r*256 + ci)
    const int K2 = 1152;  // 9 * C   (tap-major: k = r*128 + ci)
    _Float16* W1p   = (_Float16*)take((size_t)128 * K1 * 2);
    _Float16* Woffp = (_Float16*)take((size_t)128 * K2 * 2);  // M=18 padded to 128
    _Float16* Wdcnp = (_Float16*)take((size_t)128 * K2 * 2);
    _Float16* W2p   = (_Float16*)take((size_t)128 * K2 * 2);
    _Float16* Wmkp  = (_Float16*)take((size_t)128 * K2 * 2);  // M=1 padded to 128
    float*    h1    = (float*)take((size_t)BB * CC * HWP * 4);
    float*    offs  = (float*)take((size_t)BB * 18 * HWP * 4);
    _Float16* samp  = (_Float16*)take((size_t)NPIX * K2 * 2);
    float*    h2    = (float*)take((size_t)BB * CC * HWP * 4);
    float*    h3    = (float*)take((size_t)BB * CC * HWP * 4);
    float*    stats = (float*)take((size_t)2 * CC * 4);

    // ---- pack weights to f16 tap-major (pad extra M rows with zeros)
    {
        long t;
        t = (long)128 * K1;
        pack_tapmajor_kernel<<<dim3((t + 255) / 256), 256, 0, stream>>>(conv_w, W1p, 128, 128, 2 * CC);
        t = (long)128 * K2;
        pack_tapmajor_kernel<<<dim3((t + 255) / 256), 256, 0, stream>>>(off_w, Woffp, 18, 128, CC);
        pack_tapmajor_kernel<<<dim3((t + 255) / 256), 256, 0, stream>>>(dcn_w, Wdcnp, 128, 128, CC);
        pack_tapmajor_kernel<<<dim3((t + 255) / 256), 256, 0, stream>>>(conv2_w, W2p, 128, 128, CC);
        pack_tapmajor_kernel<<<dim3((t + 255) / 256), 256, 0, stream>>>(mask_w, Wmkp, 1, 128, CC);
    }

    const int nTiles = NPIX / 64;  // 392

    // ---- conv1: concat(x,y) [2C] -> h1 [C]
    conv_gemm_kernel<0><<<dim3(1, nTiles), 256, 0, stream>>>(
        x, y, /*Csplit=*/CC, nullptr, W1p, conv_b, h1,
        HH, WW, /*Co=*/CC, /*Kp=*/K1);

    // ---- offset conv: h1 -> offs [18]
    conv_gemm_kernel<0><<<dim3(1, nTiles), 256, 0, stream>>>(
        h1, h1, CC, nullptr, Woffp, off_b, offs,
        HH, WW, /*Co=*/18, K2);

    // ---- deformable bilinear sampling -> f16 patch panel
    deform_sample_kernel<<<dim3(NPIX, 9), CC, 0, stream>>>(h1, offs, samp, CC, HH, WW);

    // ---- DCN GEMM: Wdcn [C, 9C] x samp^T -> h2
    conv_gemm_kernel<1><<<dim3(1, nTiles), 256, 0, stream>>>(
        nullptr, nullptr, 0, samp, Wdcnp, dcn_b, h2,
        HH, WW, /*Co=*/CC, K2);

    // ---- conv2: h2 -> h3
    conv_gemm_kernel<0><<<dim3(1, nTiles), 256, 0, stream>>>(
        h2, h2, CC, nullptr, W2p, conv2_b, h3,
        HH, WW, /*Co=*/CC, K2);

    // ---- batchnorm stats + fused BN/GELU -> out_h
    bn_stats_kernel<<<dim3(CC), 256, 0, stream>>>(h3, stats, BB, CC, HWP);
    {
        long total = (long)BB * CC * HWP;
        bn_gelu_kernel<<<dim3((total + 255) / 256), 256, 0, stream>>>(
            h3, stats, bn_g, bn_b, out_h, CC, HWP, total);
    }

    // ---- mask conv: out_h -> out_mask [1]
    conv_gemm_kernel<0><<<dim3(1, nTiles), 256, 0, stream>>>(
        out_h, out_h, CC, nullptr, Wmkp, mask_b, out_mask,
        HH, WW, /*Co=*/1, K2);
}
